// VAE_ODE_RNN_4964982194677
// MI455X (gfx1250) — matrix operand order repaired and checked by reading, not compiled, not run on hardware
//
#include <hip/hip_runtime.h>
#include <hip/hip_bf16.h>
#include <cstddef>

typedef __bf16 bf16;
typedef bf16  v16bf  __attribute__((ext_vector_type(16)));
typedef bf16  bf16x8 __attribute__((ext_vector_type(8)));
typedef bf16  bf16x4 __attribute__((ext_vector_type(4)));
typedef float v8f    __attribute__((ext_vector_type(8)));

#define N_TRAJ 4096
#define SEQ_L  128
#define IN_DIM 32
#define LAT    64
#define UNITS  256
#define YCDIM  160      // 2*LAT + IN_DIM
#define M_BLK  32       // trajectories per workgroup
#define THREADS 256     // 8 wave32
#define NWAVES 8
#define LDC 168         // yc row stride in bf16 elems (336B = 21*16B)
#define LDH 264         // hidden row stride in bf16 elems (528B = 33*16B)

// d_ws offsets (bf16 elements); each weight stored transposed [N][K]
#define OFF_WO1T 0         // 256x64
#define OFF_WO2T 16384     // 64x256
#define OFF_WU1T 32768     // 256x160
#define OFF_WU2T 73728     // 64x256
#define OFF_WR1T 90112     // 256x160
#define OFF_WR2T 131072    // 64x256
#define OFF_WN1T 147456    // 256x160
#define OFF_WN2T 188416    // 128x256

// ---------------- weight prep: f32 [K][N] -> bf16 [N][K] ----------------
__global__ void transpose_bf16_kernel(const float* __restrict__ W,
                                      bf16* __restrict__ Wt, int K, int N) {
  int idx = blockIdx.x * blockDim.x + threadIdx.x;
  if (idx < K * N) {
    int n = idx / K;
    int k = idx - n * K;
    Wt[idx] = (bf16)W[(size_t)k * N + n];
  }
}

// ---------------- fast activations: 1 v_exp + 1 v_rcp each ----------------
__device__ __forceinline__ float fast_sigmoid(float x) {
  float e = __builtin_amdgcn_exp2f(x * -1.4426950408889634f);
  return __builtin_amdgcn_rcpf(1.0f + e);
}
__device__ __forceinline__ float fast_tanh(float x) {
  float e = __builtin_amdgcn_exp2f(x * -2.8853900817779268f);
  return (1.0f - e) * __builtin_amdgcn_rcpf(1.0f + e);
}

// ---------------- WMMA operand loads ----------------
// A tile (16x32 bf16) from LDS, row-major with stride lda.
// A k order used here (contiguous; B loads match): lane L (half h=L>>4, l16=L&15): row m = l16,
// elems 0..15 -> K = kt*32 + h*16 + 0..15
__device__ __forceinline__ v16bf load_A_tile(const bf16* A, int lda, int mt,
                                             int kt, int lane) {
  int m   = mt * 16 + (lane & 15);
  int klo = kt * 32 + ((lane >> 4) << 4);
  const bf16* p = A + m * lda + klo;
  bf16x8 lo = *(const bf16x8*)(p);
  bf16x8 hi = *(const bf16x8*)(p + 8);
  return __builtin_shufflevector(lo, hi, 0,1,2,3,4,5,6,7,8,9,10,11,12,13,14,15);
}

__device__ __forceinline__ v8f wmma_bf16(v16bf a, v16bf b, v8f c) {
  return __builtin_amdgcn_wmma_f32_16x16x32_bf16(false, a, false, b,
                                                 (short)0, c, false, false);
}

// Wide GEMM (N=256): each wave owns columns {wave*16, wave*16+128}; one A load
// feeds two WMMAs (dual-N accumulators) -> half the LDS traffic.
template <int KT, typename Epi>
__device__ __forceinline__ void gemm_n256(const bf16* A, int lda,
                                          const bf16* __restrict__ Wt, int ldk,
                                          const float* __restrict__ bias,
                                          int wave, int lane, Epi epi) {
  const int nt0 = wave, nt1 = wave + 8;
  const int l16 = lane & 15;
  const int n0 = nt0 * 16 + l16, n1 = nt1 * 16 + l16;
  const bf16* wp0 = Wt + (size_t)n0 * ldk + ((lane >> 4) << 4);
  const bf16* wp1 = Wt + (size_t)n1 * ldk + ((lane >> 4) << 4);
  const float bv0 = bias[n0], bv1 = bias[n1];
#pragma unroll
  for (int mt = 0; mt < 2; ++mt) {
    v8f acc0 = { bv0, bv0, bv0, bv0, bv0, bv0, bv0, bv0 };
    v8f acc1 = { bv1, bv1, bv1, bv1, bv1, bv1, bv1, bv1 };
#pragma unroll
    for (int kt = 0; kt < KT; ++kt) {
      v16bf a  = load_A_tile(A, lda, mt, kt, lane);
      v16bf b0 = *(const v16bf*)(wp0 + kt * 32);
      v16bf b1 = *(const v16bf*)(wp1 + kt * 32);
      acc0 = wmma_bf16(a, b0, acc0);
      acc1 = wmma_bf16(a, b1, acc1);
    }
    epi(acc0, mt, nt0);
    epi(acc1, mt, nt1);
  }
}

// Narrow GEMM (N = NT*16, NT power of 2): tiles round-robined over waves.
template <int NT, int KT, typename Epi>
__device__ __forceinline__ void gemm_block(const bf16* A, int lda,
                                           const bf16* __restrict__ Wt, int ldk,
                                           const float* __restrict__ bias,
                                           int wave, int lane, Epi epi) {
  constexpr int TILES = 2 * NT;  // 2 M-tiles (M_BLK=32)
  for (int idx = wave; idx < TILES; idx += NWAVES) {
    int nt = idx & (NT - 1);
    int mt = idx / NT;
    int n  = nt * 16 + (lane & 15);
    float bv = bias[n];
    v8f acc = { bv, bv, bv, bv, bv, bv, bv, bv };
    const bf16* wp = Wt + (size_t)n * ldk + ((lane >> 4) << 4);
#pragma unroll
    for (int kt = 0; kt < KT; ++kt) {
      v16bf a = load_A_tile(A, lda, mt, kt, lane);
      v16bf b = *(const v16bf*)(wp + kt * 32);
      acc = wmma_bf16(a, b, acc);
    }
    epi(acc, mt, nt);
  }
}

// ---------------- main scan kernel ----------------
__global__ __launch_bounds__(THREADS)
void ode_rnn_scan(const float* __restrict__ data, const float* __restrict__ ts,
                  const bf16* __restrict__ ws,
                  const float* __restrict__ bo1, const float* __restrict__ bo2,
                  const float* __restrict__ bu1, const float* __restrict__ bu2,
                  const float* __restrict__ br1, const float* __restrict__ br2,
                  const float* __restrict__ bn1, const float* __restrict__ bn2,
                  float* __restrict__ out) {
  __shared__ __attribute__((aligned(16))) float sy[M_BLK * LAT];     // y_mean
  __shared__ __attribute__((aligned(16))) float slv[M_BLK * LAT];    // y_logvar
  __shared__ __attribute__((aligned(16))) float syode[M_BLK * LAT];  // y after ODE
  __shared__ __attribute__((aligned(16))) float supd[M_BLK * LAT];   // update gate
  __shared__ __attribute__((aligned(16))) bf16  syc[M_BLK * LDC];    // [y|lv|x] / cc
  __shared__ __attribute__((aligned(16))) bf16  sh [M_BLK * LDH];    // hidden 256

  const int tid  = threadIdx.x;
  const int lane = tid & 31;
  const int wave = tid >> 5;
  const int traj0 = blockIdx.x * M_BLK;

  const bf16* Wo1t = ws + OFF_WO1T;
  const bf16* Wo2t = ws + OFF_WO2T;
  const bf16* Wu1t = ws + OFF_WU1T;
  const bf16* Wu2t = ws + OFF_WU2T;
  const bf16* Wr1t = ws + OFF_WR1T;
  const bf16* Wr2t = ws + OFF_WR2T;
  const bf16* Wn1t = ws + OFF_WN1T;
  const bf16* Wn2t = ws + OFF_WN2T;

  for (int i = tid; i < M_BLK * LAT; i += THREADS) { sy[i] = 0.f; slv[i] = 0.f; }
  __syncthreads();

  // per-thread x-slice base (16B per step)
  const int xm = tid >> 3, xj = (tid & 7) * 4;
  const float* xbase = data + ((size_t)(traj0 + xm)) * SEQ_L * IN_DIM + xj;

  // shared epilogue: tanh -> hidden buffer
  auto epi_tanh = [&](v8f acc, int mt, int nt) {
    int n  = nt * 16 + (lane & 15);
    int mb = mt * 16 + ((lane >> 4) << 3);
#pragma unroll
    for (int r = 0; r < 8; ++r)
      sh[(mb + r) * LDH + n] = (bf16)fast_tanh(acc[r]);
  };

#pragma unroll 1
  for (int t = 0; t < SEQ_L - 1; ++t) {
    float dt = (t == 0) ? (ts[1] - ts[0]) : (ts[t] - ts[t + 1]);

    // P0: stage yc = [bf16(y) | bf16(lv) | bf16(x_t)], vectorized 16B stores
    {
      int b = tid * 8;            // 2048 elems over 256 threads
      int m = b >> 6, k = b & 63;
      const float4 y0 = *(const float4*)&sy [m * LAT + k];
      const float4 y1 = *(const float4*)&sy [m * LAT + k + 4];
      const float4 v0 = *(const float4*)&slv[m * LAT + k];
      const float4 v1 = *(const float4*)&slv[m * LAT + k + 4];
      bf16x8 py = { (bf16)y0.x, (bf16)y0.y, (bf16)y0.z, (bf16)y0.w,
                    (bf16)y1.x, (bf16)y1.y, (bf16)y1.z, (bf16)y1.w };
      bf16x8 pv = { (bf16)v0.x, (bf16)v0.y, (bf16)v0.z, (bf16)v0.w,
                    (bf16)v1.x, (bf16)v1.y, (bf16)v1.z, (bf16)v1.w };
      *(bf16x8*)&syc[m * LDC + k]      = py;
      *(bf16x8*)&syc[m * LDC + 64 + k] = pv;

      const float4 xv = *(const float4*)(xbase + (size_t)(t + 1) * IN_DIM);
      bf16x4 px = { (bf16)xv.x, (bf16)xv.y, (bf16)xv.z, (bf16)xv.w };
      *(bf16x4*)&syc[xm * LDC + 128 + xj] = px;

      // prefetch next step's x slice into near caches while this step computes
      if (t + 2 < SEQ_L)
        __builtin_prefetch(xbase + (size_t)(t + 2) * IN_DIM, 0, 1);
    }
    __syncthreads();

    // P1: ODE layer 1: tanh(y @ Wo1 + bo1) -> sh   (K=64)
    gemm_n256<2>(syc, LDC, Wo1t, LAT, bo1, wave, lane, epi_tanh);
    __syncthreads();

    // P2: ODE layer 2: yode = y + dt*(sh @ Wo2 + bo2); refresh yc[:,0:64]
    gemm_block<4, 8>(sh, LDH, Wo2t, UNITS, bo2, wave, lane,
      [&](v8f acc, int mt, int nt) {
        int n  = nt * 16 + (lane & 15);
        int mb = mt * 16 + ((lane >> 4) << 3);
#pragma unroll
        for (int r = 0; r < 8; ++r) {
          int m = mb + r;
          float yo = sy[m * LAT + n] + dt * acc[r];
          syode[m * LAT + n] = yo;
          syc[m * LDC + n]   = (bf16)yo;
        }
      });
    __syncthreads();

    // P3/P4: update gate = sigmoid(mlp(yc; Wu))
    gemm_n256<5>(syc, LDC, Wu1t, YCDIM, bu1, wave, lane, epi_tanh);
    __syncthreads();
    gemm_block<4, 8>(sh, LDH, Wu2t, UNITS, bu2, wave, lane,
      [&](v8f acc, int mt, int nt) {
        int n  = nt * 16 + (lane & 15);
        int mb = mt * 16 + ((lane >> 4) << 3);
#pragma unroll
        for (int r = 0; r < 8; ++r)
          supd[(mb + r) * LAT + n] = fast_sigmoid(acc[r]);
      });
    __syncthreads();

    // P5/P6: reset gate, fused into cc = [yode*r | lv*r | x] (x kept in place)
    gemm_n256<5>(syc, LDC, Wr1t, YCDIM, br1, wave, lane, epi_tanh);
    __syncthreads();
    gemm_block<4, 8>(sh, LDH, Wr2t, UNITS, br2, wave, lane,
      [&](v8f acc, int mt, int nt) {
        int n  = nt * 16 + (lane & 15);
        int mb = mt * 16 + ((lane >> 4) << 3);
#pragma unroll
        for (int r = 0; r < 8; ++r) {
          int m = mb + r;
          float rr = fast_sigmoid(acc[r]);
          syc[m * LDC + n]      = (bf16)(syode[m * LAT + n] * rr);
          syc[m * LDC + 64 + n] = (bf16)(slv  [m * LAT + n] * rr);
        }
      });
    __syncthreads();

    // P7/P8: candidate h = mlp(cc; Wn); blend new state with update gate
    gemm_n256<5>(syc, LDC, Wn1t, YCDIM, bn1, wave, lane, epi_tanh);
    __syncthreads();
    gemm_block<8, 8>(sh, LDH, Wn2t, UNITS, bn2, wave, lane,
      [&](v8f acc, int mt, int nt) {
        int n  = nt * 16 + (lane & 15);
        int mb = mt * 16 + ((lane >> 4) << 3);
        if (n < 64) {             // uniform per wave (nt uniform)
#pragma unroll
          for (int r = 0; r < 8; ++r) {
            int m = mb + r;
            float u = supd[m * LAT + n];
            sy[m * LAT + n] = (1.f - u) * acc[r] + u * syode[m * LAT + n];
          }
        } else {
          int n2 = n - 64;
#pragma unroll
          for (int r = 0; r < 8; ++r) {
            int m = mb + r;
            float u = supd[m * LAT + n2];
            slv[m * LAT + n2] = (1.f - u) * fabsf(acc[r]) + u * slv[m * LAT + n2];
          }
        }
      });
    __syncthreads();
  }

  // write final (yi, yi_logvar) concatenated
  for (int i = tid; i < M_BLK * LAT; i += THREADS) {
    int m = i >> 6, n = i & 63;
    out[(size_t)(traj0 + m) * LAT + n] = sy[i];
    out[(size_t)N_TRAJ * LAT + (size_t)(traj0 + m) * LAT + n] = slv[i];
  }
}

// ---------------- launch ----------------
extern "C" void kernel_launch(void* const* d_in, const int* in_sizes, int n_in,
                              void* d_out, int out_size, void* d_ws, size_t ws_size,
                              hipStream_t stream) {
  const float* data = (const float*)d_in[0];
  const float* ts   = (const float*)d_in[1];
  const float* Wo1  = (const float*)d_in[2];
  const float* bo1  = (const float*)d_in[3];
  const float* Wo2  = (const float*)d_in[4];
  const float* bo2  = (const float*)d_in[5];
  const float* Wu1  = (const float*)d_in[6];
  const float* bu1  = (const float*)d_in[7];
  const float* Wu2  = (const float*)d_in[8];
  const float* bu2  = (const float*)d_in[9];
  const float* Wr1  = (const float*)d_in[10];
  const float* br1  = (const float*)d_in[11];
  const float* Wr2  = (const float*)d_in[12];
  const float* br2  = (const float*)d_in[13];
  const float* Wn1  = (const float*)d_in[14];
  const float* bn1  = (const float*)d_in[15];
  const float* Wn2  = (const float*)d_in[16];
  const float* bn2  = (const float*)d_in[17];
  bf16* ws = (bf16*)d_ws;

  auto tr = [&](const float* W, int off, int K, int N) {
    int n = K * N;
    transpose_bf16_kernel<<<(n + 255) / 256, 256, 0, stream>>>(W, ws + off, K, N);
  };
  tr(Wo1, OFF_WO1T, LAT,   UNITS);
  tr(Wo2, OFF_WO2T, UNITS, LAT);
  tr(Wu1, OFF_WU1T, YCDIM, UNITS);
  tr(Wu2, OFF_WU2T, UNITS, LAT);
  tr(Wr1, OFF_WR1T, YCDIM, UNITS);
  tr(Wr2, OFF_WR2T, UNITS, LAT);
  tr(Wn1, OFF_WN1T, YCDIM, UNITS);
  tr(Wn2, OFF_WN2T, UNITS, 2 * LAT);

  ode_rnn_scan<<<N_TRAJ / M_BLK, THREADS, 0, stream>>>(
      data, ts, ws, bo1, bo2, bu1, bu2, br1, br2, bn1, bn2, (float*)d_out);
}
